// MMHSA_81853486727199
// MI455X (gfx1250) — matrix operand order repaired and checked by reading, not compiled, not run on hardware
//
#include <hip/hip_runtime.h>

// ---------------------------------------------------------------------------
// Matryoshka multi-head GQA attention for MI455X (gfx1250, wave32, WMMA).
//
// Strategy (compute-bound, ~60 GFLOP on ~100MB traffic => push WMMA):
//   1) cast x -> bf16; gather-cast weights with the reorder permutations
//      folded in (Q/K/V: column gather; Wo: row gather via out[rev]@Wo ==
//      out@Wo[fwd]); gather biases.
//   2) Q/K/V projections via a generic bf16 WMMA GEMM (f32 accumulate).
//   3) Flash-attention with 3 nested softmax levels (matryoshka), causal
//      mask computed analytically, v_wmma_f32_16x16x32_bf16 for QK^T and AV.
//      Softmax row reductions use DPP ROW_ROR butterflies (pure VALU, no
//      LDS round-trips) since the reduction domain is a 16-lane DPP row.
//   4) Output projection WMMA GEMM (f32 out) + bo.
//
// Workspace requirement: ~51 MB.
// ---------------------------------------------------------------------------

typedef __attribute__((ext_vector_type(16))) __bf16 v16bf;
typedef __attribute__((ext_vector_type(8)))  float  v8f;

#define NEGBIG (-1.0e30f)

__device__ __forceinline__ unsigned short f2bfbits(float f) {
    unsigned int x = __builtin_bit_cast(unsigned int, f);
    unsigned int r = (x + 0x7FFFu + ((x >> 16) & 1u)) >> 16;   // RNE
    return (unsigned short)r;
}
__device__ __forceinline__ __bf16 bfbits(unsigned short u) {
    return __builtin_bit_cast(__bf16, u);
}

// DPP lane-rotate within 16-lane rows: ROW_ROR:n ctrl = 0x120 | n
template <int CTRL>
__device__ __forceinline__ float dpp_rot(float x) {
    return __builtin_bit_cast(float, __builtin_amdgcn_update_dpp(
        0, __builtin_bit_cast(int, x), CTRL, 0xF, 0xF, true));
}
// All-lanes reduction over each 16-lane row (butterfly by rotation).
__device__ __forceinline__ float row16_max(float x) {
    x = fmaxf(x, dpp_rot<0x128>(x));   // ROW_ROR:8
    x = fmaxf(x, dpp_rot<0x124>(x));   // ROW_ROR:4
    x = fmaxf(x, dpp_rot<0x122>(x));   // ROW_ROR:2
    x = fmaxf(x, dpp_rot<0x121>(x));   // ROW_ROR:1
    return x;
}
__device__ __forceinline__ float row16_sum(float x) {
    x += dpp_rot<0x128>(x);
    x += dpp_rot<0x124>(x);
    x += dpp_rot<0x122>(x);
    x += dpp_rot<0x121>(x);
    return x;
}

// forward matryoshka reorder index: position i (head-major h*128+d) -> source col
__device__ __forceinline__ int mfwd(int i, int nh) {
    int h = i >> 7, d = i & 127;
    if (d < 32) return (h << 5) + d;
    if (d < 64) return (nh << 5) + (h << 5) + (d - 32);
    return (nh << 6) + (h << 6) + (d - 64);
}

// A-fragment K index for 16-bit A 16x32 (contiguous, matches B): lanes<16 hold K {0..15}
__device__ __forceinline__ int kmapA(int e, int half) {
    return e + (half << 4);
}

// ------------------------------ preprocessing ------------------------------

__global__ __launch_bounds__(256) void cast_x_kernel(const float* __restrict__ x,
                                                     unsigned short* __restrict__ y, int n) {
    int i = blockIdx.x * 256 + threadIdx.x;
    if (i < n) y[i] = f2bfbits(x[i]);
}

__global__ __launch_bounds__(256) void gather_w_cols_kernel(const float* __restrict__ W,
                                                            unsigned short* __restrict__ Wb,
                                                            int rows, int ncols, int nh) {
    int i = blockIdx.x * 256 + threadIdx.x;
    if (i < rows * ncols) {
        int r = i / ncols, c = i - r * ncols;
        Wb[i] = f2bfbits(W[(size_t)r * ncols + mfwd(c, nh)]);
    }
}

__global__ __launch_bounds__(256) void gather_wo_rows_kernel(const float* __restrict__ W,
                                                             unsigned short* __restrict__ Wb) {
    int i = blockIdx.x * 256 + threadIdx.x;
    if (i < 2048 * 2048) {
        int r = i >> 11, c = i & 2047;
        Wb[i] = f2bfbits(W[(size_t)mfwd(r, 16) * 2048 + c]);
    }
}

__global__ __launch_bounds__(256) void gather_bias_kernel(const float* __restrict__ b,
                                                          float* __restrict__ bg, int n, int nh) {
    int i = blockIdx.x * 256 + threadIdx.x;
    if (i < n) bg[i] = b[mfwd(i, nh)];
}

// ------------------------------ generic WMMA GEMM --------------------------
// C[M,N] = A[M,K](bf16) @ B[K,N](bf16) + bias[N], block tile 128x64, BK=32.

template <bool OUT_BF16>
__global__ __launch_bounds__(256) void gemm_bf16_kernel(const unsigned short* __restrict__ A,
                                                        const unsigned short* __restrict__ Bm,
                                                        const float* __restrict__ bias,
                                                        void* __restrict__ Cout,
                                                        int M, int N, int K) {
    __shared__ __align__(16) unsigned short sA[128 * 32];
    __shared__ __align__(16) unsigned short sB[32 * 64];
    const int tid = threadIdx.x, lane = tid & 31, wave = tid >> 5;
    const int half = lane >> 4, ln = lane & 15;
    const int col0 = blockIdx.x * 64, row0 = blockIdx.y * 128;

    v8f acc[4];
#pragma unroll
    for (int t = 0; t < 4; ++t)
#pragma unroll
        for (int r = 0; r < 8; ++r) acc[t][r] = 0.f;

    for (int kc = 0; kc < K; kc += 32) {
#pragma unroll
        for (int i = 0; i < 2; ++i) {
            int idx = tid + i * 256;          // 512 chunks of 8 bf16
            int r = idx >> 2, cb = idx & 3;
            *(uint4*)&sA[r * 32 + cb * 8] =
                *(const uint4*)&A[(size_t)(row0 + r) * K + kc + cb * 8];
        }
        {
            int r = tid >> 3, cb = tid & 7;
            *(uint4*)&sB[r * 64 + cb * 8] =
                *(const uint4*)&Bm[(size_t)(kc + r) * N + col0 + cb * 8];
        }
        if (kc + 32 < K)  // gfx1250 global_prefetch_b8 on next A chunk
            __builtin_prefetch(&A[(size_t)(row0 + (tid >> 1)) * K + kc + 32], 0, 1);
        __syncthreads();

        v16bf a;
#pragma unroll
        for (int e = 0; e < 16; ++e)
            a[e] = bfbits(sA[(wave * 16 + ln) * 32 + kmapA(e, half)]);
#pragma unroll
        for (int t = 0; t < 4; ++t) {
            v16bf bb;
#pragma unroll
            for (int e = 0; e < 16; ++e)
                bb[e] = bfbits(sB[(half * 16 + e) * 64 + t * 16 + ln]);
            acc[t] = __builtin_amdgcn_wmma_f32_16x16x32_bf16(
                false, a, false, bb, (short)0, acc[t], false, false);
        }
        __syncthreads();
    }

#pragma unroll
    for (int t = 0; t < 4; ++t)
#pragma unroll
        for (int r = 0; r < 8; ++r) {
            int row = row0 + wave * 16 + r + half * 8;
            int col = col0 + t * 16 + ln;
            float v = acc[t][r] + bias[col];
            if (OUT_BF16)
                ((unsigned short*)Cout)[(size_t)row * N + col] = f2bfbits(v);
            else
                ((float*)Cout)[(size_t)row * N + col] = v;
        }
}

// ------------------------------ flash matryoshka attention -----------------

template <int NT>
__device__ __forceinline__ void level_update(v8f s0, v8f s1, float scale,
                                             float* mstate, float* lstate, v8f* ot,
                                             int qrow0, int jbase, int lane,
                                             const unsigned short* Vp, int voff,
                                             unsigned short* ldsP) {
    const int half = lane >> 4, ln = lane & 15;
    float fact[8];
#pragma unroll
    for (int r = 0; r < 8; ++r) {
        int row = qrow0 + r + (half << 3);
        int c0 = jbase + ln, c1 = c0 + 16;
        bool ok0 = (c0 <= row), ok1 = (c1 <= row);
        float x0 = s0[r] * scale, x1 = s1[r] * scale;
        float rm = row16_max(fmaxf(ok0 ? x0 : NEGBIG, ok1 ? x1 : NEGBIG));
        float mnew = fmaxf(mstate[r], rm);
        float p0 = ok0 ? __expf(x0 - mnew) : 0.f;
        float p1 = ok1 ? __expf(x1 - mnew) : 0.f;
        float rs = row16_sum(p0 + p1);
        float f = __expf(mstate[r] - mnew);
        lstate[r] = lstate[r] * f + rs;
        mstate[r] = mnew;
        fact[r]   = f;
        int rowl = r + (half << 3);
        ldsP[rowl * 32 + ln]      = f2bfbits(p0);
        ldsP[rowl * 32 + 16 + ln] = f2bfbits(p1);
    }
#pragma unroll
    for (int t = 0; t < NT; ++t)
#pragma unroll
        for (int r = 0; r < 8; ++r) ot[t][r] *= fact[r];

    v16bf pa;  // D-layout -> A-layout via per-wave LDS bounce
#pragma unroll
    for (int e = 0; e < 16; ++e)
        pa[e] = bfbits(ldsP[ln * 32 + kmapA(e, half)]);

#pragma unroll
    for (int t = 0; t < NT; ++t) {
        v16bf vb;
#pragma unroll
        for (int e = 0; e < 16; ++e) {
            int key = jbase + (half << 4) + e;
            vb[e] = bfbits(Vp[(size_t)key * 512 + voff + t * 16 + ln]);
        }
        ot[t] = __builtin_amdgcn_wmma_f32_16x16x32_bf16(
            false, pa, false, vb, (short)0, ot[t], false, false);
    }
}

__global__ __launch_bounds__(256) void attn_kernel(const unsigned short* __restrict__ Qb,
                                                   const unsigned short* __restrict__ Kb,
                                                   const unsigned short* __restrict__ Vb,
                                                   unsigned short* __restrict__ Ob) {
    __shared__ unsigned short ldsP[8][16 * 32];
    const int lane = threadIdx.x & 31, wave = threadIdx.x >> 5;
    const int half = lane >> 4, ln = lane & 15;
    const int qc = blockIdx.x & 7;          // S/128 = 8
    const int h  = (blockIdx.x >> 3) & 15;
    const int b  = blockIdx.x >> 7;
    const int g  = h >> 2;                  // kv group (repeat=4, consecutive)
    const int qrow0 = qc * 128 + wave * 16;

    const unsigned short* Qp = Qb + (size_t)(b * 1024) * 2048 + h * 128;
    const unsigned short* Kp = Kb + (size_t)(b * 1024) * 512 + g * 128;
    const unsigned short* Vp = Vb + (size_t)(b * 1024) * 512 + g * 128;

    v16bf a[4];  // q fragment, 4 chunks of 32 dims
#pragma unroll
    for (int c = 0; c < 4; ++c)
#pragma unroll
        for (int e = 0; e < 16; ++e)
            a[c][e] = bfbits(Qp[(size_t)(qrow0 + ln) * 2048 + c * 32 + kmapA(e, half)]);

    float m0[8], l0[8], m1[8], l1[8], m2[8], l2[8];
    v8f o0[2], o1[2], o2[4];
#pragma unroll
    for (int r = 0; r < 8; ++r) {
        m0[r] = m1[r] = m2[r] = NEGBIG;
        l0[r] = l1[r] = l2[r] = 0.f;
    }
#pragma unroll
    for (int t = 0; t < 2; ++t)
#pragma unroll
        for (int r = 0; r < 8; ++r) { o0[t][r] = 0.f; o1[t][r] = 0.f; }
#pragma unroll
    for (int t = 0; t < 4; ++t)
#pragma unroll
        for (int r = 0; r < 8; ++r) o2[t][r] = 0.f;

    const int jend = qrow0 + 15;
    for (int jbase = 0; jbase <= jend; jbase += 32) {
        v8f t0, t1;
#pragma unroll
        for (int r = 0; r < 8; ++r) { t0[r] = 0.f; t1[r] = 0.f; }

#pragma unroll
        for (int c = 0; c < 4; ++c) {
            v16bf b0, b1;  // K^T fragments: lane=key col, element=K dim
#pragma unroll
            for (int e = 0; e < 16; ++e) {
                int kd = c * 32 + half * 16 + e;
                b0[e] = bfbits(Kp[(size_t)(jbase + ln) * 512 + kd]);
                b1[e] = bfbits(Kp[(size_t)(jbase + 16 + ln) * 512 + kd]);
            }
            t0 = __builtin_amdgcn_wmma_f32_16x16x32_bf16(false, a[c], false, b0,
                                                         (short)0, t0, false, false);
            t1 = __builtin_amdgcn_wmma_f32_16x16x32_bf16(false, a[c], false, b1,
                                                         (short)0, t1, false, false);
            if (c == 0)
                level_update<2>(t0, t1, 0.17677669529663687f, m0, l0, o0,
                                qrow0, jbase, lane, Vp, 0, ldsP[wave]);
            if (c == 1)
                level_update<2>(t0, t1, 0.125f, m1, l1, o1,
                                qrow0, jbase, lane, Vp, 32, ldsP[wave]);
        }
        level_update<4>(t0, t1, 0.08838834764831843f, m2, l2, o2,
                        qrow0, jbase, lane, Vp, 64, ldsP[wave]);
    }

    unsigned short* Op = Ob + (size_t)(b * 1024) * 2048 + h * 128;
#pragma unroll
    for (int t = 0; t < 2; ++t)
#pragma unroll
        for (int r = 0; r < 8; ++r) {
            int row = qrow0 + r + half * 8;
            Op[(size_t)row * 2048 + 0  + t * 16 + ln] = f2bfbits(o0[t][r] / l0[r]);
            Op[(size_t)row * 2048 + 32 + t * 16 + ln] = f2bfbits(o1[t][r] / l1[r]);
        }
#pragma unroll
    for (int t = 0; t < 4; ++t)
#pragma unroll
        for (int r = 0; r < 8; ++r) {
            int row = qrow0 + r + half * 8;
            Op[(size_t)row * 2048 + 64 + t * 16 + ln] = f2bfbits(o2[t][r] / l2[r]);
        }
}

// ------------------------------ launch -------------------------------------

extern "C" void kernel_launch(void* const* d_in, const int* in_sizes, int n_in,
                              void* d_out, int out_size, void* d_ws, size_t ws_size,
                              hipStream_t stream) {
    const float* x  = (const float*)d_in[0];
    // d_in[1] = mask (unused: causal mask derived analytically)
    const float* Wq = (const float*)d_in[2];
    const float* bq = (const float*)d_in[3];
    const float* Wk = (const float*)d_in[4];
    const float* bk = (const float*)d_in[5];
    const float* Wv = (const float*)d_in[6];
    const float* bv = (const float*)d_in[7];
    const float* Wo = (const float*)d_in[8];
    const float* bo = (const float*)d_in[9];
    float* out = (float*)d_out;

    const int B = 2, S = 1024, D = 2048, T = B * S;  // T = 2048 tokens
    char* w = (char*)d_ws;
    size_t off = 0;
    auto alloc = [&](size_t bytes) {
        void* p = w + off;
        off = (off + bytes + 255) & ~(size_t)255;
        return p;
    };
    unsigned short* xb  = (unsigned short*)alloc((size_t)T * D * 2);
    unsigned short* WqB = (unsigned short*)alloc((size_t)D * 2048 * 2);
    unsigned short* WkB = (unsigned short*)alloc((size_t)D * 512 * 2);
    unsigned short* WvB = (unsigned short*)alloc((size_t)D * 512 * 2);
    unsigned short* WoB = (unsigned short*)alloc((size_t)2048 * D * 2);
    unsigned short* Qb  = (unsigned short*)alloc((size_t)T * 2048 * 2);
    unsigned short* Kb  = (unsigned short*)alloc((size_t)T * 512 * 2);
    unsigned short* Vb  = (unsigned short*)alloc((size_t)T * 512 * 2);
    unsigned short* Ob  = (unsigned short*)alloc((size_t)T * 2048 * 2);
    float* bqG = (float*)alloc(2048 * 4);
    float* bkG = (float*)alloc(512 * 4);
    float* bvG = (float*)alloc(512 * 4);
    (void)ws_size;  // requires ~51 MB

    // 1) preprocessing
    {
        int n = T * D;
        cast_x_kernel<<<(n + 255) / 256, 256, 0, stream>>>(x, xb, n);
    }
    gather_w_cols_kernel<<<(D * 2048) / 256, 256, 0, stream>>>(Wq, WqB, D, 2048, 16);
    gather_w_cols_kernel<<<(D * 512) / 256, 256, 0, stream>>>(Wk, WkB, D, 512, 4);
    gather_w_cols_kernel<<<(D * 512) / 256, 256, 0, stream>>>(Wv, WvB, D, 512, 4);
    gather_wo_rows_kernel<<<(2048 * 2048) / 256, 256, 0, stream>>>(Wo, WoB);
    gather_bias_kernel<<<2048 / 256, 256, 0, stream>>>(bq, bqG, 2048, 16);
    gather_bias_kernel<<<2, 256, 0, stream>>>(bk, bkG, 512, 4);
    gather_bias_kernel<<<2, 256, 0, stream>>>(bv, bvG, 512, 4);

    // 2) projections (bf16 out)
    gemm_bf16_kernel<true><<<dim3(2048 / 64, T / 128), 256, 0, stream>>>(
        xb, WqB, bqG, Qb, T, 2048, D);
    gemm_bf16_kernel<true><<<dim3(512 / 64, T / 128), 256, 0, stream>>>(
        xb, WkB, bkG, Kb, T, 512, D);
    gemm_bf16_kernel<true><<<dim3(512 / 64, T / 128), 256, 0, stream>>>(
        xb, WvB, bvG, Vb, T, 512, D);

    // 3) matryoshka flash attention: grid = B*H*(S/128) = 256 blocks
    attn_kernel<<<256, 256, 0, stream>>>(Qb, Kb, Vb, Ob);

    // 4) output projection (f32 out)
    gemm_bf16_kernel<false><<<dim3(2048 / 64, T / 128), 256, 0, stream>>>(
        Ob, WoB, bo, out, T, 2048, D);

    (void)in_sizes; (void)n_in; (void)out_size;
}
